// Attention_25091198943598
// MI455X (gfx1250) — matrix operand order repaired and checked by reading, not compiled, not run on hardware
//
#include <hip/hip_runtime.h>

#define DIM      1024
#define N_HEADS  16
#define N_KV     4
#define HEAD     64
#define BSZ      2
#define SEQ      2048
#define MROWS    (BSZ * SEQ)   // 4096

typedef __attribute__((ext_vector_type(16))) __bf16 v16bf;
typedef __attribute__((ext_vector_type(8)))  float  v8f;
typedef __attribute__((ext_vector_type(4)))  unsigned int v4u;
typedef __attribute__((ext_vector_type(4)))  float  v4f;

union Frag {
    v16bf v;
    v4u q[2];
    unsigned short u[16];
};

__device__ __forceinline__ unsigned short f2bf(float f) {
    union { float f; unsigned int u; } a; a.f = f;
    unsigned int u = a.u;
    u += 0x7FFFu + ((u >> 16) & 1u);   // round-to-nearest-even
    return (unsigned short)(u >> 16);
}

// packed pair conversion: (a -> low 16, b -> high 16)
__device__ __forceinline__ unsigned int f2bf2(float a, float b) {
#if __has_builtin(__builtin_amdgcn_cvt_pk_bf16_f32)
    auto r = __builtin_amdgcn_cvt_pk_bf16_f32(a, b);
    union { decltype(r) v; unsigned int u; } c; c.v = r;
    return c.u;
#else
    return (unsigned int)f2bf(a) | ((unsigned int)f2bf(b) << 16);
#endif
}

__device__ __forceinline__ void wait_asynccnt0() {
#if __has_builtin(__builtin_amdgcn_s_wait_asynccnt)
    __builtin_amdgcn_s_wait_asynccnt(0);
#else
    asm volatile("s_wait_asynccnt 0x0" ::: "memory");
#endif
}

__device__ __forceinline__ v8f wmma_bf16(const Frag& a, const Frag& b, v8f c) {
    return __builtin_amdgcn_wmma_f32_16x16x32_bf16(
        false, a.v, false, b.v, (short)0, c, false, false);
}

// ---------------------------------------------------------------------------
// Generic tiled GEMM: C(MxN) = A(MxK) @ B(KxN)
//   AMODE: 0 = A fp32 row-major (convert through VGPRs),
//          1 = A bf16 row-major (async global->LDS copy, no VGPR staging)
//   OMODE: 0 = C bf16 row-major, 1 = C bf16 transposed (Ct[N][M]), 2 = C fp32
// Tile 128x128, K-step 32, 256 threads = 8 waves (wave tile 32x64).
// Double-buffered LDS: 1 barrier/iter, next-tile loads overlap WMMA.
// ---------------------------------------------------------------------------
template <int AMODE, int OMODE>
__global__ __launch_bounds__(256)
void gemm_wmma(const void* __restrict__ Av, const float* __restrict__ B,
               void* __restrict__ Cv, int M, int N, int K)
{
    __shared__ unsigned short As[2][128][32];   // bf16 [m][k]
    __shared__ unsigned short Bs[2][128][32];   // bf16 transposed [n][k]

    const int tid    = threadIdx.x;
    const int tile_m = blockIdx.y * 128;
    const int tile_n = blockIdx.x * 128;
    const int wid    = tid >> 5;
    const int lane   = tid & 31;
    const int lhalf  = lane >> 4;
    const int l15    = lane & 15;
    const int wm     = (wid & 3) * 32;
    const int wn     = (wid >> 2) * 64;

    // B staging: each thread owns a 4(k) x 4(n) block
    const int kb4 = (tid >> 5) * 4;        // k base 0..28
    const int c4B = (tid & 31) * 4;        // n base 0..124

    v8f acc[2][4];
#pragma unroll
    for (int i = 0; i < 2; ++i)
#pragma unroll
        for (int j = 0; j < 4; ++j) acc[i][j] = (v8f)(0.0f);

    v4f aF[4];     // A fp32 staging
    v4f bF[4];     // B staging (4 consecutive k rows, same 4 n cols)

    const int nt = K / 32;

    // A bf16 path: async copy straight into LDS (ASYNCcnt-tracked)
    auto async_loadA = [&](int k0, int bufT) {
        const unsigned short* A = (const unsigned short*)Av;
#pragma unroll
        for (int i = 0; i < 2; ++i) {
            int slot = tid + i * 256;
            int row  = slot >> 2;
            int c8   = (slot & 3) * 8;
            const unsigned short* g = A + (size_t)(tile_m + row) * K + k0 + c8;
            unsigned lds = (unsigned)(size_t)&As[bufT][row][c8];  // flat->LDS: low 32 bits
            unsigned long long ga = (unsigned long long)(size_t)g;
            asm volatile("global_load_async_to_lds_b128 %0, %1, off"
                         :: "v"(lds), "v"(ga) : "memory");
        }
    };

    auto load_tile = [&](int k0) {
        if (AMODE == 0) {
            const float* A = (const float*)Av;
#pragma unroll
            for (int i = 0; i < 4; ++i) {
                int slot = tid + i * 256;
                int row  = slot >> 3;
                int c4   = (slot & 7) * 4;
                aF[i] = *(const v4f*)(A + (size_t)(tile_m + row) * K + k0 + c4);
            }
        }
#pragma unroll
        for (int j = 0; j < 4; ++j)
            bF[j] = *(const v4f*)(B + (size_t)(k0 + kb4 + j) * N + tile_n + c4B);
    };

    auto store_tile = [&](int buf) {
        if (AMODE == 0) {
#pragma unroll
            for (int i = 0; i < 4; ++i) {
                int slot = tid + i * 256;
                int row  = slot >> 3;
                int c4   = (slot & 7) * 4;
                *(unsigned int*)&As[buf][row][c4]     = f2bf2(aF[i].x, aF[i].y);
                *(unsigned int*)&As[buf][row][c4 + 2] = f2bf2(aF[i].z, aF[i].w);
            }
        }
        // transpose 4x4 block: pack along k
#pragma unroll
        for (int cc = 0; cc < 4; ++cc) {
            *(unsigned int*)&Bs[buf][c4B + cc][kb4]     = f2bf2(bF[0][cc], bF[1][cc]);
            *(unsigned int*)&Bs[buf][c4B + cc][kb4 + 2] = f2bf2(bF[2][cc], bF[3][cc]);
        }
    };

    // prologue: fill buffer 0
    if (AMODE == 1) async_loadA(0, 0);
    load_tile(0);
    store_tile(0);
    if (AMODE == 1) wait_asynccnt0();
    int buf = 0;

    for (int it = 0; it < nt; ++it) {
        __syncthreads();   // publishes buf; frees buf^1 for refill
        if (it + 1 < nt) {
            if (AMODE == 1) async_loadA((it + 1) * 32, buf ^ 1);
            load_tile((it + 1) * 32);
            if (it + 2 < nt) {   // prefetch tile after next into cache
                int kp = (it + 2) * 32;
                if (AMODE == 0)
                    __builtin_prefetch((const float*)Av + (size_t)(tile_m + (tid >> 3)) * K + kp, 0, 0);
                else
                    __builtin_prefetch((const unsigned short*)Av + (size_t)(tile_m + (tid >> 2)) * K + kp, 0, 0);
                __builtin_prefetch(B + (size_t)(kp + kb4) * N + tile_n + c4B, 0, 0);
            }
        }

        // ---- fragments + WMMA from buffer `buf` ----
        Frag a[2], b[4];
#pragma unroll
        for (int im = 0; im < 2; ++im) {
            int row = wm + im * 16 + l15;
            int kb  = lhalf * 8;
            a[im].q[0] = *(const v4u*)&As[buf][row][kb];
            a[im].q[1] = *(const v4u*)&As[buf][row][kb + 16];
        }
#pragma unroll
        for (int in = 0; in < 4; ++in) {
            int col = wn + in * 16 + l15;
            int kb  = lhalf * 8;
            b[in].q[0] = *(const v4u*)&Bs[buf][col][kb];
            b[in].q[1] = *(const v4u*)&Bs[buf][col][kb + 16];
        }
#pragma unroll
        for (int im = 0; im < 2; ++im)
#pragma unroll
            for (int in = 0; in < 4; ++in)
                acc[im][in] = wmma_bf16(a[im], b[in], acc[im][in]);

        if (it + 1 < nt) {
            store_tile(buf ^ 1);
            if (AMODE == 1) wait_asynccnt0();  // async data must land before next barrier
        }
        buf ^= 1;
    }

    // ---- epilogue ----
    if (OMODE == 0) {
        unsigned short* C = (unsigned short*)Cv;
#pragma unroll
        for (int im = 0; im < 2; ++im)
#pragma unroll
            for (int in = 0; in < 4; ++in)
#pragma unroll
                for (int r = 0; r < 8; ++r) {
                    int row = tile_m + wm + im * 16 + r + 8 * lhalf;
                    int col = tile_n + wn + in * 16 + l15;
                    C[(size_t)row * N + col] = f2bf(acc[im][in][r]);
                }
    } else if (OMODE == 1) {
        unsigned short* C = (unsigned short*)Cv;   // Ct[N][M]
#pragma unroll
        for (int im = 0; im < 2; ++im)
#pragma unroll
            for (int in = 0; in < 4; ++in) {
                int col = tile_n + wn + in * 16 + l15;
                int m0  = tile_m + wm + im * 16 + 8 * lhalf;
                union { unsigned int w[4]; v4u q; } pk;
#pragma unroll
                for (int p = 0; p < 4; ++p)
                    pk.w[p] = f2bf2(acc[im][in][2 * p], acc[im][in][2 * p + 1]);
                *(v4u*)(C + (size_t)col * M + m0) = pk.q;
            }
    } else {
        float* C = (float*)Cv;
#pragma unroll
        for (int im = 0; im < 2; ++im)
#pragma unroll
            for (int in = 0; in < 4; ++in)
#pragma unroll
                for (int r = 0; r < 8; ++r) {
                    int row = tile_m + wm + im * 16 + r + 8 * lhalf;
                    int col = tile_n + wn + in * 16 + l15;
                    C[(size_t)row * N + col] = acc[im][in][r];
                }
    }
}

// ---------------------------------------------------------------------------
// Flash-style attention.  Grid: (SEQ/128, N_HEADS, BSZ), 256 threads.
// Each wave owns 16 query rows; keys processed in chunks of 64 with online
// softmax (amortizes shuffle reductions over 4 score tiles).
// Q [t][H*64] bf16, K [t][KV*64] bf16, Vt [KV*64][MROWS] bf16.
// ---------------------------------------------------------------------------
__global__ __launch_bounds__(256)
void attn_wmma(const unsigned short* __restrict__ Q,
               const unsigned short* __restrict__ Kp,
               const unsigned short* __restrict__ Vt,
               unsigned short* __restrict__ AO)
{
    __shared__ unsigned short Pl[8][16][64];   // per-wave P staging (C->A relayout)

    const int tid   = threadIdx.x;
    const int wid   = tid >> 5;
    const int lane  = tid & 31;
    const int lhalf = lane >> 4;
    const int l15   = lane & 15;
    const int b     = blockIdx.z;
    const int h     = blockIdx.y;
    const int kvh   = h >> 2;                  // GQA: head -> kv head
    const int q0    = blockIdx.x * 128 + wid * 16;

    // Q A-fragments (16 rows x 64 head dims = two K=32 chunks)
    Frag aq[2];
    {
        const unsigned short* qrow =
            Q + (size_t)(b * SEQ + q0 + l15) * DIM + h * HEAD;
#pragma unroll
        for (int c = 0; c < 2; ++c) {
            int kb = c * 32 + lhalf * 8;
            aq[c].q[0] = *(const v4u*)(qrow + kb);
            aq[c].q[1] = *(const v4u*)(qrow + kb + 16);
        }
    }

    v8f o[4];
#pragma unroll
    for (int in = 0; in < 4; ++in) o[in] = (v8f)(0.0f);
    float mrow[8], lrow[8];
#pragma unroll
    for (int r = 0; r < 8; ++r) { mrow[r] = -1e30f; lrow[r] = 0.0f; }

    const float scale = 0.125f;                // 1/sqrt(64)

    const unsigned short* kbase = Kp + (size_t)b * SEQ * (N_KV * HEAD) + kvh * HEAD;
    const unsigned short* vbase = Vt + (size_t)(kvh * HEAD) * MROWS + b * SEQ;

    for (int j = 0; j < SEQ; j += 64) {
        // ---- scores: four 16x16 tiles, head-dim 64 accumulated over 2 WMMAs ----
        v8f s[4];
#pragma unroll
        for (int t = 0; t < 4; ++t) {
            Frag bk[2];
            const unsigned short* krow =
                kbase + (size_t)(j + t * 16 + l15) * (N_KV * HEAD);
#pragma unroll
            for (int c = 0; c < 2; ++c) {
                int kb = c * 32 + lhalf * 8;
                bk[c].q[0] = *(const v4u*)(krow + kb);
                bk[c].q[1] = *(const v4u*)(krow + kb + 16);
            }
            v8f z = (v8f)(0.0f);
            z = wmma_bf16(aq[0], bk[0], z);
            z = wmma_bf16(aq[1], bk[1], z);
            s[t] = z;
        }

        // ---- online softmax (rows live across 16-lane half groups) ----
        float corr[8];
#pragma unroll
        for (int r = 0; r < 8; ++r) {
            float mx = fmaxf(fmaxf(s[0][r], s[1][r]), fmaxf(s[2][r], s[3][r]));
#pragma unroll
            for (int sh = 1; sh < 16; sh <<= 1)
                mx = fmaxf(mx, __shfl_xor(mx, sh, 16));
            mx *= scale;
            float mnew = fmaxf(mrow[r], mx);
            corr[r] = __expf(mrow[r] - mnew);
            mrow[r] = mnew;
            float psum = 0.0f;
#pragma unroll
            for (int t = 0; t < 4; ++t) {
                float p = __expf(s[t][r] * scale - mnew);
                s[t][r] = p;
                psum += p;
            }
#pragma unroll
            for (int sh = 1; sh < 16; sh <<= 1)
                psum += __shfl_xor(psum, sh, 16);
            lrow[r] = lrow[r] * corr[r] + psum;
        }
#pragma unroll
        for (int in = 0; in < 4; ++in)
#pragma unroll
            for (int r = 0; r < 8; ++r) o[in][r] *= corr[r];

        // ---- P: C-layout -> LDS -> A-fragment layout (16x64) ----
#pragma unroll
        for (int t = 0; t < 4; ++t)
#pragma unroll
            for (int r = 0; r < 8; ++r)
                Pl[wid][r + 8 * lhalf][t * 16 + l15] = f2bf(s[t][r]);
        Frag pa[2];
        {
            const unsigned short* prow = &Pl[wid][l15][0];
#pragma unroll
            for (int c = 0; c < 2; ++c) {
                int kb = c * 32 + lhalf * 8;
                pa[c].q[0] = *(const v4u*)(prow + kb);
                pa[c].q[1] = *(const v4u*)(prow + kb + 16);
            }
        }

        // ---- O += P @ V  (keys contiguous in transposed V) ----
#pragma unroll
        for (int in = 0; in < 4; ++in) {
            const unsigned short* vrow =
                vbase + (size_t)(in * 16 + l15) * MROWS + j + lhalf * 8;
            Frag bv0, bv1;
            bv0.q[0] = *(const v4u*)(vrow);
            bv0.q[1] = *(const v4u*)(vrow + 16);
            bv1.q[0] = *(const v4u*)(vrow + 32);
            bv1.q[1] = *(const v4u*)(vrow + 48);
            o[in] = wmma_bf16(pa[0], bv0, o[in]);
            o[in] = wmma_bf16(pa[1], bv1, o[in]);
        }
    }

    // ---- finalize: O /= l, store bf16 ----
#pragma unroll
    for (int r = 0; r < 8; ++r) lrow[r] = 1.0f / lrow[r];
#pragma unroll
    for (int in = 0; in < 4; ++in)
#pragma unroll
        for (int r = 0; r < 8; ++r) {
            int qr = q0 + r + 8 * lhalf;
            AO[(size_t)(b * SEQ + qr) * DIM + h * HEAD + in * 16 + l15] =
                f2bf(o[in][r] * lrow[r]);
        }
}

// ---------------------------------------------------------------------------
extern "C" void kernel_launch(void* const* d_in, const int* in_sizes, int n_in,
                              void* d_out, int out_size, void* d_ws, size_t ws_size,
                              hipStream_t stream)
{
    const float* x  = (const float*)d_in[0];
    const float* wq = (const float*)d_in[1];
    const float* wk = (const float*)d_in[2];
    const float* wv = (const float*)d_in[3];
    const float* wo = (const float*)d_in[4];

    // workspace layout (bf16): Q | K | Vt | attn_out  (20 MiB total)
    unsigned short* Qws = (unsigned short*)d_ws;
    unsigned short* Kws = Qws + (size_t)MROWS * DIM;
    unsigned short* Vtw = Kws + (size_t)MROWS * (N_KV * HEAD);
    unsigned short* AOw = Vtw + (size_t)(N_KV * HEAD) * MROWS;

    dim3 blk(256);
    // QKV projections (fp32 in, bf16 out; V stored transposed)
    gemm_wmma<0, 0><<<dim3(DIM / 128, MROWS / 128), blk, 0, stream>>>(
        x, wq, Qws, MROWS, DIM, DIM);
    gemm_wmma<0, 0><<<dim3((N_KV * HEAD) / 128, MROWS / 128), blk, 0, stream>>>(
        x, wk, Kws, MROWS, N_KV * HEAD, DIM);
    gemm_wmma<0, 1><<<dim3((N_KV * HEAD) / 128, MROWS / 128), blk, 0, stream>>>(
        x, wv, Vtw, MROWS, N_KV * HEAD, DIM);
    // attention
    attn_wmma<<<dim3(SEQ / 128, N_HEADS, BSZ), blk, 0, stream>>>(
        Qws, Kws, Vtw, AOw);
    // output projection (bf16 A via async global->LDS, fp32 out)
    gemm_wmma<1, 2><<<dim3(DIM / 128, MROWS / 128), blk, 0, stream>>>(
        AOw, wo, d_out, MROWS, DIM, DIM);
}
